// PhyloGNN_61280593379791
// MI455X (gfx1250) — matrix operand order repaired and checked by reading, not compiled, not run on hardware
//
#include <hip/hip_runtime.h>

#define N_NODES 100000
#define EMB     128
#define NLAYERS 3
#define N_EDGES 1600000

typedef __attribute__((ext_vector_type(16))) __bf16 v16bf;
typedef __attribute__((ext_vector_type(8)))  float  v8f;

// ---------------------------------------------------------------------------
// Zero the message buffer (vectorized)
// ---------------------------------------------------------------------------
__global__ void zero_f4(float4* __restrict__ p, int n4) {
    int i = blockIdx.x * blockDim.x + threadIdx.x;
    if (i < n4) p[i] = make_float4(0.f, 0.f, 0.f, 0.f);
}

// ---------------------------------------------------------------------------
// Split W into transposed bf16 hi/lo:  whiT/wloT layout [layer][n][k]
// ---------------------------------------------------------------------------
__global__ void prep_w(const float* __restrict__ W,
                       __bf16* __restrict__ whiT, __bf16* __restrict__ wloT) {
    int i = blockIdx.x * blockDim.x + threadIdx.x;   // over NLAYERS*EMB*EMB
    if (i >= NLAYERS * EMB * EMB) return;
    int l = i / (EMB * EMB);
    int r = i % (EMB * EMB);
    int k = r / EMB;          // source row    (K)
    int n = r % EMB;          // source column (N)
    float w = W[i];
    __bf16 hi = (__bf16)w;
    __bf16 lo = (__bf16)(w - (float)hi);
    size_t o = (size_t)l * EMB * EMB + (size_t)n * EMB + k;   // transposed
    whiT[o] = hi;
    wloT[o] = lo;
}

// ---------------------------------------------------------------------------
// Edge scatter-add: one wave (32 lanes) per edge; lane moves 4 floats.
// m and x both fit in 192MB L2, so gathers + f32 atomics stay near-chip.
// ---------------------------------------------------------------------------
__global__ void scatter_add(const int* __restrict__ src, const int* __restrict__ dst,
                            const float* __restrict__ x, float* __restrict__ m) {
    int e    = (blockIdx.x * blockDim.x + threadIdx.x) >> 5;
    int lane = threadIdx.x & 31;
    if (e >= N_EDGES) return;
    int s = src[e];
    int d = dst[e];
    float4 v = ((const float4*)(x + (size_t)s * EMB))[lane];
    float* mp = m + (size_t)d * EMB + lane * 4;
    atomicAdd(mp + 0, v.x);
    atomicAdd(mp + 1, v.y);
    atomicAdd(mp + 2, v.z);
    atomicAdd(mp + 3, v.w);
}

// ---------------------------------------------------------------------------
// x_out = x_in + gelu( (m * deg_inv) @ W + b )
// 16 rows per block; 8 waves each own a 16-col tile; bf16x3 WMMA, f32 acc.
// A-tile hi/lo split is done ONCE cooperatively into bf16 LDS tiles, so the
// per-wave inner loop is pure ds_load -> v_wmma (no conversion VALU).
// ---------------------------------------------------------------------------
#define APAD 8   // bf16 elements of row padding: row stride 272B -> conflict-free
__launch_bounds__(256)
__global__ void gemm_gelu_res(const float* __restrict__ m,
                              const float* __restrict__ deg_inv,
                              const __bf16* __restrict__ whiT,
                              const __bf16* __restrict__ wloT,
                              const float* __restrict__ bias,
                              const float* __restrict__ x_in,
                              float* __restrict__ x_out) {
    __shared__ __bf16 Ahi_s[16][EMB + APAD];
    __shared__ __bf16 Alo_s[16][EMB + APAD];
    const int tid     = threadIdx.x;
    const int rowBase = blockIdx.x * 16;

    // cooperative load of the 16x128 A tile; fuse deg_inv scale + hi/lo split
#pragma unroll
    for (int i = 0; i < 8; ++i) {
        int lin = tid + i * 256;
        int rr  = lin >> 7;
        int cc  = lin & 127;
        float a = m[(size_t)(rowBase + rr) * EMB + cc] * deg_inv[rowBase + rr];
        __bf16 hi = (__bf16)a;
        Ahi_s[rr][cc] = hi;
        Alo_s[rr][cc] = (__bf16)(a - (float)hi);
    }
    __syncthreads();

    const int wave = tid >> 5;
    const int lane = tid & 31;
    const int half = lane >> 4;       // 0: lanes 0-15, 1: lanes 16-31
    const int l16  = lane & 15;
    const int n0   = wave * 16;       // this wave's column tile

    v8f acc = {0.f, 0.f, 0.f, 0.f, 0.f, 0.f, 0.f, 0.f};

#pragma unroll
    for (int ks = 0; ks < 4; ++ks) {          // K = 128 in 4 steps of 32
        const int kbase = ks * 32;
        v16bf Ahi, Alo, Bhi, Blo;
        const int khalf = half * 8;           // A: lanes 16-31 hold K+8 block
#pragma unroll
        for (int r = 0; r < 8; ++r) {
            // --- A fragment (16-bit A 16x32 layout, ISA 7.12.2); pairs are
            //     contiguous bf16 in LDS -> merges into ds_load_b128 ---
            int k0 = ((r < 4) ? (2 * r) : (16 + 2 * (r - 4))) + khalf;
            Ahi[2 * r]     = Ahi_s[l16][kbase + k0];
            Ahi[2 * r + 1] = Ahi_s[l16][kbase + k0 + 1];
            Alo[2 * r]     = Alo_s[l16][kbase + k0];
            Alo[2 * r + 1] = Alo_s[l16][kbase + k0 + 1];
            // --- B fragment: lane=N, VGPR r holds K pair; same K order as A
            int kb = kbase + ((r < 4) ? (2 * r) : (16 + 2 * (r - 4))) + half * 8;
            size_t wo = (size_t)(n0 + l16) * EMB + kb;   // transposed [n][k]
            Bhi[2 * r]     = whiT[wo];
            Bhi[2 * r + 1] = whiT[wo + 1];
            Blo[2 * r]     = wloT[wo];
            Blo[2 * r + 1] = wloT[wo + 1];
        }
        // bf16x3 split-precision accumulation (~f32 accuracy, f32 accumulator)
        acc = __builtin_amdgcn_wmma_f32_16x16x32_bf16(false, Ahi, false, Bhi, (short)0, acc, false, false);
        acc = __builtin_amdgcn_wmma_f32_16x16x32_bf16(false, Alo, false, Bhi, (short)0, acc, false, false);
        acc = __builtin_amdgcn_wmma_f32_16x16x32_bf16(false, Ahi, false, Blo, (short)0, acc, false, false);
    }

    // C/D layout: lane = N, VGPR r -> row r + 8*half. Fuse bias + GELU + residual.
    const int col  = n0 + l16;
    const float bc = bias[col];
#pragma unroll
    for (int r = 0; r < 8; ++r) {
        int row  = rowBase + r + 8 * half;
        float v  = acc[r] + bc;
        float g  = 0.5f * v * (1.0f + erff(v * 0.70710678118654752f));  // exact gelu
        size_t o = (size_t)row * EMB + col;
        x_out[o] = x_in[o] + g;
    }
}

// ---------------------------------------------------------------------------
// LayerNorm over the 128-feature dim: one wave32 per node.
// ---------------------------------------------------------------------------
__global__ void layernorm_k(const float* __restrict__ x,
                            const float* __restrict__ gamma,
                            const float* __restrict__ beta,
                            float* __restrict__ out) {
    int node = (blockIdx.x * blockDim.x + threadIdx.x) >> 5;
    int lane = threadIdx.x & 31;
    if (node >= N_NODES) return;
    float4 v = ((const float4*)(x + (size_t)node * EMB))[lane];
    float s  = v.x + v.y + v.z + v.w;
    float ss = v.x * v.x + v.y * v.y + v.z * v.z + v.w * v.w;
#pragma unroll
    for (int off = 16; off >= 1; off >>= 1) {
        s  += __shfl_xor(s,  off, 32);
        ss += __shfl_xor(ss, off, 32);
    }
    float mu   = s * (1.0f / EMB);
    float var  = ss * (1.0f / EMB) - mu * mu;
    float rstd = rsqrtf(var + 1e-5f);
    float4 g  = ((const float4*)gamma)[lane];
    float4 bb = ((const float4*)beta)[lane];
    float4 o;
    o.x = (v.x - mu) * rstd * g.x + bb.x;
    o.y = (v.y - mu) * rstd * g.y + bb.y;
    o.z = (v.z - mu) * rstd * g.z + bb.z;
    o.w = (v.w - mu) * rstd * g.w + bb.w;
    ((float4*)out)[(size_t)node * (EMB / 4) + lane] = o;
}

// ---------------------------------------------------------------------------
extern "C" void kernel_launch(void* const* d_in, const int* in_sizes, int n_in,
                              void* d_out, int out_size, void* d_ws, size_t ws_size,
                              hipStream_t stream) {
    const int*   edges    = (const int*)d_in[0];       // [2, N_EDGES]
    const int*   src      = edges;
    const int*   dst      = edges + N_EDGES;
    const float* deg_inv  = (const float*)d_in[1];     // [N]
    const float* node_emb = (const float*)d_in[2];     // [N, 128]
    const float* W        = (const float*)d_in[3];     // [3, 128, 128]
    const float* bias     = (const float*)d_in[4];     // [3, 128]
    const float* gamma    = (const float*)d_in[5];     // [128]
    const float* beta     = (const float*)d_in[6];     // [128]
    float*       out      = (float*)d_out;             // [N, 128]

    // workspace: x (51.2MB) | m (51.2MB) | bf16 W hi/lo transposed (~200KB)
    float*  ws_x = (float*)d_ws;
    float*  ws_m = ws_x + (size_t)N_NODES * EMB;
    __bf16* whiT = (__bf16*)(ws_m + (size_t)N_NODES * EMB);
    __bf16* wloT = whiT + (size_t)NLAYERS * EMB * EMB;

    prep_w<<<(NLAYERS * EMB * EMB + 255) / 256, 256, 0, stream>>>(W, whiT, wloT);

    for (int l = 0; l < NLAYERS; ++l) {
        const float* xin = (l == 0) ? node_emb : ws_x;

        zero_f4<<<(N_NODES * EMB / 4 + 255) / 256, 256, 0, stream>>>(
            (float4*)ws_m, N_NODES * EMB / 4);

        scatter_add<<<(N_EDGES + 7) / 8, 256, 0, stream>>>(src, dst, xin, ws_m);

        gemm_gelu_res<<<N_NODES / 16, 256, 0, stream>>>(
            ws_m, deg_inv,
            whiT + (size_t)l * EMB * EMB, wloT + (size_t)l * EMB * EMB,
            bias + (size_t)l * EMB, xin, ws_x);
    }

    layernorm_k<<<(N_NODES + 7) / 8, 256, 0, stream>>>(ws_x, gamma, beta, out);
}
